// MPGNN_43954695308055
// MI455X (gfx1250) — matrix operand order repaired and checked by reading, not compiled, not run on hardware
//
#include <hip/hip_runtime.h>

// ---------------------------------------------------------------------------
// MPGNN inference for MI455X (gfx1250, wave32).
// bf16 WMMA (v_wmma_f32_16x16x32_bf16) GEMMs, f32 accumulate, f32 global state.
// ---------------------------------------------------------------------------

typedef __bf16 bf16;
typedef __attribute__((ext_vector_type(16))) __bf16 v16bf;
typedef __attribute__((ext_vector_type(8)))  __bf16 v8bf;
typedef __attribute__((ext_vector_type(8)))  float   v8f;

#define WPB 2            // wave (=16-row tile) count per block
#define LN_EPS 1e-5f

union V16u { v16bf v; v8bf h2[2]; bf16 e[16]; };

// monotonic uint mapping of f32 for atomicMax
static __device__ __forceinline__ unsigned fmap(float f) {
  unsigned u = __float_as_uint(f);
  return (u >> 31) ? ~u : (u | 0x80000000u);
}
static __device__ __forceinline__ float funmap(unsigned u) {
  unsigned b = (u >> 31) ? (u & 0x7fffffffu) : ~u;
  return __uint_as_float(b);
}
static __device__ __forceinline__ float silu(float x) {
  return x / (1.0f + __expf(-x));
}

// one GEMM accumulation pass: A (16 x kp, LDS bf16 row-major) x B (packed
// fragments) -> 16x64 tile in c[4] (each v_wmma_f32_16x16x32_bf16)
static __device__ __forceinline__ void wmma_pass(const bf16* A, int kp, int nch,
                                                 const v16bf* __restrict__ B,
                                                 v8f c[4]) {
  const int lane = threadIdx.x & 31;
  const int m = lane & 15, hf = lane >> 4;
  for (int kc = 0; kc < nch; ++kc) {
    V16u a;
    const bf16* rp = A + m * kp + kc * 32 + 16 * hf;  // K run: base..+15
    a.h2[0] = *(const v8bf*)rp;
    a.h2[1] = *(const v8bf*)(rp + 8);
#pragma unroll
    for (int t = 0; t < 4; ++t) {
      v16bf b = B[(kc * 4 + t) * 32 + lane];
      c[t] = __builtin_amdgcn_wmma_f32_16x16x32_bf16(false, a.v, false, b,
                                                     (short)0, c[t], false, false);
    }
  }
}

// add bias, store C tile as f32 [16][64] to LDS scratch
static __device__ __forceinline__ void c_bias_to_S(const v8f c[4],
                                                   const float* __restrict__ bi,
                                                   float* S) {
  const int lane = threadIdx.x & 31;
  const int m = lane & 15, hf = lane >> 4;
#pragma unroll
  for (int t = 0; t < 4; ++t) {
    const int nn = m + 16 * t;
    const float bb = bi[nn];
#pragma unroll
    for (int v = 0; v < 8; ++v) S[(v + 8 * hf) * 64 + nn] = c[t][v] + bb;
  }
}

// lanes 0..15: per-row LayerNorm + SiLU, write bf16 row to A1
static __device__ __forceinline__ void ln_silu_to_A(const float* S,
                                                    const float* __restrict__ g,
                                                    const float* __restrict__ be,
                                                    bf16* A1) {
  const int lane = threadIdx.x & 31;
  if (lane < 16) {
    const float* r = S + lane * 64;
    float mu = 0.0f;
    for (int j = 0; j < 64; ++j) mu += r[j];
    mu *= (1.0f / 64.0f);
    float var = 0.0f;
    for (int j = 0; j < 64; ++j) { float d = r[j] - mu; var += d * d; }
    var *= (1.0f / 64.0f);
    const float rs = rsqrtf(var + LN_EPS);
    bf16* w = A1 + lane * 64;
    for (int j = 0; j < 64; ++j) {
      float hn = (r[j] - mu) * rs * g[j] + be[j];
      w[j] = (bf16)silu(hn);
    }
  }
}

// hidden trunk of an MLP_block: [lin+LN+SiLU] (x2 if W1 != null); result in A1
static __device__ __forceinline__ void mlp_hidden(const bf16* A0, int kp,
    bf16* A1, float* S,
    const v16bf* W0, const float* bi0, const float* g0, const float* be0,
    const v16bf* W1, const float* bi1, const float* g1, const float* be1) {
  v8f c[4] = {{}, {}, {}, {}};
  wmma_pass(A0, kp, kp >> 5, W0, c);
  c_bias_to_S(c, bi0, S);
  __syncthreads();
  ln_silu_to_A(S, g0, be0, A1);
  __syncthreads();
  if (W1 != nullptr) {
    v8f d[4] = {{}, {}, {}, {}};
    wmma_pass(A1, 64, 2, W1, d);
    c_bias_to_S(d, bi1, S);
    __syncthreads();
    ln_silu_to_A(S, g1, be1, A1);
    __syncthreads();
  }
}

// ---------------------------------------------------------------------------
// weight packer: W f32 [K][64] -> B-fragment bf16 layout, k contiguous per lane half
// fragment f = kc*4+t ; lane holds N=(lane&15)+16t, K = kc*32 + 16*(lane>>4)+e
// ---------------------------------------------------------------------------
__global__ void pack_kernel(const float* __restrict__ W, int fanin,
                            v16bf* __restrict__ dst) {
  const int f = blockIdx.x, lane = threadIdx.x;
  const int kc = f >> 2, t = f & 3;
  const int m = lane & 15, hf = lane >> 4;
  V16u v;
#pragma unroll
  for (int e = 0; e < 16; ++e) {
    const int K = kc * 32 + 16 * hf + e;
    const int N = m + 16 * t;
    v.e[e] = (bf16)((K < fanin) ? W[(size_t)K * 64 + N] : 0.0f);
  }
  dst[f * 32 + lane] = v.v;
}

// global embedding MLP (2 -> 16 -> 16), trivially small: 1 thread
__global__ void glob_kernel(const float* __restrict__ u,
                            const float* W0, const float* b0,
                            const float* g, const float* be,
                            const float* Wo, const float* bo,
                            float* __restrict__ ug) {
  if (threadIdx.x != 0 || blockIdx.x != 0) return;
  float hb[16], a[16];
  for (int j = 0; j < 16; ++j) hb[j] = u[0] * W0[j] + u[1] * W0[16 + j] + b0[j];
  float mu = 0.0f;
  for (int j = 0; j < 16; ++j) mu += hb[j];
  mu *= (1.0f / 16.0f);
  float var = 0.0f;
  for (int j = 0; j < 16; ++j) { float d = hb[j] - mu; var += d * d; }
  var *= (1.0f / 16.0f);
  const float rs = rsqrtf(var + LN_EPS);
  for (int j = 0; j < 16; ++j) a[j] = silu((hb[j] - mu) * rs * g[j] + be[j]);
  for (int j = 0; j < 16; ++j) {
    float s = bo[j];
    for (int i = 0; i < 16; ++i) s += a[i] * Wo[i * 16 + j];
    ug[j] = s;
  }
}

__global__ void init_agg_kernel(float* __restrict__ sumb,
                                unsigned* __restrict__ mxb,
                                float* __restrict__ cntb, int N) {
  const int i = blockIdx.x * blockDim.x + threadIdx.x;
  if (i < N * 64) { sumb[i] = 0.0f; mxb[i] = 0u; }
  if (i < N) cntb[i] = 0.0f;
}

// ---------------------------------------------------------------------------
// node embedding: x [N,7] -> h [N,64]   (MLP: 7->64, LN, SiLU, 64->64)
// ---------------------------------------------------------------------------
__global__ __launch_bounds__(WPB * 32) void node_emb_kernel(
    const float* __restrict__ x, int N, float* __restrict__ hbuf,
    const v16bf* W0, const float* bi0, const float* g0, const float* be0,
    const v16bf* Wo, const float* bio) {
  __shared__ __align__(32) bf16 A0s[WPB][16 * 32];
  __shared__ __align__(32) bf16 A1s[WPB][16 * 64];
  __shared__ float Ss[WPB][16 * 64];
  const int wave = threadIdx.x >> 5, lane = threadIdx.x & 31;
  const int m = lane & 15, hf = lane >> 4;
  const int tile = blockIdx.x * WPB + wave;
  const bool active = tile * 16 < N;
  const int n0 = tile * 16;
  bf16* A0 = A0s[wave]; bf16* A1 = A1s[wave]; float* S = Ss[wave];
  if (active && hf == 0) {
    const float* xp = x + (size_t)(n0 + m) * 7;
    bf16* row = A0 + m * 32;
    for (int j = 0; j < 7; ++j)  row[j] = (bf16)xp[j];
    for (int j = 7; j < 32; ++j) row[j] = (bf16)0.0f;
  }
  __syncthreads();
  mlp_hidden(A0, 32, A1, S, W0, bi0, g0, be0, nullptr, nullptr, nullptr, nullptr);
  v8f o[4] = {{}, {}, {}, {}};
  wmma_pass(A1, 64, 2, Wo, o);
  if (active) {
#pragma unroll
    for (int t = 0; t < 4; ++t) {
      const int nn = m + 16 * t; const float bb = bio[nn];
#pragma unroll
      for (int v = 0; v < 8; ++v)
        hbuf[(size_t)(n0 + v + 8 * hf) * 64 + nn] = o[t][v] + bb;
    }
  }
}

// ---------------------------------------------------------------------------
// edge embedding: r_ij = pos[row]-pos[col] -> edge_attr [E,64]
// ---------------------------------------------------------------------------
__global__ __launch_bounds__(WPB * 32) void edge_emb_kernel(
    const int* __restrict__ ei, int E, const float* __restrict__ pos,
    float* __restrict__ ea,
    const v16bf* W0, const float* bi0, const float* g0, const float* be0,
    const v16bf* Wo, const float* bio) {
  __shared__ __align__(32) bf16 A0s[WPB][16 * 32];
  __shared__ __align__(32) bf16 A1s[WPB][16 * 64];
  __shared__ float Ss[WPB][16 * 64];
  const int wave = threadIdx.x >> 5, lane = threadIdx.x & 31;
  const int m = lane & 15, hf = lane >> 4;
  const int tile = blockIdx.x * WPB + wave;
  const bool active = tile * 16 < E;
  const int e0 = tile * 16;
  bf16* A0 = A0s[wave]; bf16* A1 = A1s[wave]; float* S = Ss[wave];
  if (active && hf == 0) {
    const int e = e0 + m;
    const int r = ei[e], cI = ei[E + e];
    bf16* row = A0 + m * 32;
    for (int k = 0; k < 3; ++k)
      row[k] = (bf16)(pos[(size_t)r * 3 + k] - pos[(size_t)cI * 3 + k]);
    for (int j = 3; j < 32; ++j) row[j] = (bf16)0.0f;
  }
  __syncthreads();
  mlp_hidden(A0, 32, A1, S, W0, bi0, g0, be0, nullptr, nullptr, nullptr, nullptr);
  v8f o[4] = {{}, {}, {}, {}};
  wmma_pass(A1, 64, 2, Wo, o);
  if (active) {
#pragma unroll
    for (int t = 0; t < 4; ++t) {
      const int nn = m + 16 * t; const float bb = bio[nn];
#pragma unroll
      for (int v = 0; v < 8; ++v)
        ea[(size_t)(e0 + v + 8 * hf) * 64 + nn] = o[t][v] + bb;
    }
  }
}

// ---------------------------------------------------------------------------
// edge model: cat(h[row], h[col], edge_attr, ug) [208 -> pad 224] -> MLP
// writes new edge_attr in place + scatter sum/max/count per receiver
// ---------------------------------------------------------------------------
__global__ __launch_bounds__(WPB * 32) void edge_mlp_kernel(
    const int* __restrict__ ei, int E,
    const float* __restrict__ hbuf, float* __restrict__ ea,
    const float* __restrict__ ug,
    const v16bf* W0, const float* bi0, const float* g0, const float* be0,
    const v16bf* W1, const float* bi1, const float* g1, const float* be1,
    const v16bf* Wo, const float* bio,
    float* __restrict__ sumb, unsigned* __restrict__ mxb,
    float* __restrict__ cntb) {
  __shared__ __align__(32) bf16 A0s[WPB][16 * 224];
  __shared__ __align__(32) bf16 A1s[WPB][16 * 64];
  __shared__ float Ss[WPB][16 * 64];
  __shared__ int colv[WPB][16];
  const int wave = threadIdx.x >> 5, lane = threadIdx.x & 31;
  const int m = lane & 15, hf = lane >> 4;
  const int tile = blockIdx.x * WPB + wave;
  const bool active = tile * 16 < E;
  const int e0 = tile * 16;
  bf16* A0 = A0s[wave]; bf16* A1 = A1s[wave]; float* S = Ss[wave];
  if (active) {
    const int e = e0 + m;
    const int r = ei[e], cI = ei[E + e];
    if (hf == 0) { colv[wave][m] = cI; atomicAdd(&cntb[cI], 1.0f); }
    const float4* hr = (const float4*)(hbuf + (size_t)r * 64 + hf * 32);
    const float4* hc = (const float4*)(hbuf + (size_t)cI * 64 + hf * 32);
    const float4* ep = (const float4*)(ea + (size_t)e * 64 + hf * 32);
    bf16* row = A0 + m * 224;
#pragma unroll
    for (int q = 0; q < 8; ++q) {
      const int j = hf * 32 + q * 4;
      float4 a4 = hr[q];
      row[j] = (bf16)a4.x; row[j + 1] = (bf16)a4.y;
      row[j + 2] = (bf16)a4.z; row[j + 3] = (bf16)a4.w;
      float4 b4 = hc[q];
      row[64 + j] = (bf16)b4.x; row[64 + j + 1] = (bf16)b4.y;
      row[64 + j + 2] = (bf16)b4.z; row[64 + j + 3] = (bf16)b4.w;
      float4 c4 = ep[q];
      row[128 + j] = (bf16)c4.x; row[128 + j + 1] = (bf16)c4.y;
      row[128 + j + 2] = (bf16)c4.z; row[128 + j + 3] = (bf16)c4.w;
    }
    if (hf == 0) { for (int j = 0; j < 16; ++j) row[192 + j] = (bf16)ug[j]; }
    else         { for (int j = 0; j < 16; ++j) row[208 + j] = (bf16)0.0f; }
  }
  __syncthreads();
  mlp_hidden(A0, 224, A1, S, W0, bi0, g0, be0, W1, bi1, g1, be1);
  v8f o[4] = {{}, {}, {}, {}};
  wmma_pass(A1, 64, 2, Wo, o);
  if (active) {
#pragma unroll
    for (int t = 0; t < 4; ++t) {
      const int nn = m + 16 * t; const float bb = bio[nn];
#pragma unroll
      for (int v = 0; v < 8; ++v) {
        const int M = v + 8 * hf;
        const float val = o[t][v] + bb;
        ea[(size_t)(e0 + M) * 64 + nn] = val;
        const int cn = colv[wave][M];
        atomicAdd(&sumb[(size_t)cn * 64 + nn], val);
        atomicMax(&mxb[(size_t)cn * 64 + nn], fmap(val));
      }
    }
  }
}

// ---------------------------------------------------------------------------
// node model: cat(sum, max, mean, h, ug) [272 -> pad 288] -> MLP -> h in place
// ---------------------------------------------------------------------------
__global__ __launch_bounds__(WPB * 32) void node_mlp_kernel(
    int N, float* __restrict__ hbuf,
    const float* __restrict__ sumb, const unsigned* __restrict__ mxb,
    const float* __restrict__ cntb, const float* __restrict__ ug,
    const v16bf* W0, const float* bi0, const float* g0, const float* be0,
    const v16bf* W1, const float* bi1, const float* g1, const float* be1,
    const v16bf* Wo, const float* bio) {
  __shared__ __align__(32) bf16 A0s[WPB][16 * 288];
  __shared__ __align__(32) bf16 A1s[WPB][16 * 64];
  __shared__ float Ss[WPB][16 * 64];
  const int wave = threadIdx.x >> 5, lane = threadIdx.x & 31;
  const int m = lane & 15, hf = lane >> 4;
  const int tile = blockIdx.x * WPB + wave;
  const bool active = tile * 16 < N;
  const int n0 = tile * 16;
  bf16* A0 = A0s[wave]; bf16* A1 = A1s[wave]; float* S = Ss[wave];
  if (active) {
    const int n = n0 + m;
    const float cf = cntb[n];
    const float inv = 1.0f / fmaxf(cf, 1.0f);
    const bool has = cf > 0.0f;
    const float4* sp = (const float4*)(sumb + (size_t)n * 64 + hf * 32);
    const uint4*  mp = (const uint4*)(mxb + (size_t)n * 64 + hf * 32);
    const float4* hp = (const float4*)(hbuf + (size_t)n * 64 + hf * 32);
    bf16* row = A0 + m * 288;
#pragma unroll
    for (int q = 0; q < 8; ++q) {
      const int j = hf * 32 + q * 4;
      float4 s4 = sp[q]; uint4 u4 = mp[q]; float4 h4 = hp[q];
      row[j] = (bf16)s4.x; row[j + 1] = (bf16)s4.y;
      row[j + 2] = (bf16)s4.z; row[j + 3] = (bf16)s4.w;
      row[64 + j]     = (bf16)(has ? funmap(u4.x) : 0.0f);
      row[64 + j + 1] = (bf16)(has ? funmap(u4.y) : 0.0f);
      row[64 + j + 2] = (bf16)(has ? funmap(u4.z) : 0.0f);
      row[64 + j + 3] = (bf16)(has ? funmap(u4.w) : 0.0f);
      row[128 + j] = (bf16)(s4.x * inv); row[128 + j + 1] = (bf16)(s4.y * inv);
      row[128 + j + 2] = (bf16)(s4.z * inv); row[128 + j + 3] = (bf16)(s4.w * inv);
      row[192 + j] = (bf16)h4.x; row[192 + j + 1] = (bf16)h4.y;
      row[192 + j + 2] = (bf16)h4.z; row[192 + j + 3] = (bf16)h4.w;
    }
    if (hf == 0) { for (int j = 0; j < 16; ++j) row[256 + j] = (bf16)ug[j]; }
    else         { for (int j = 0; j < 16; ++j) row[272 + j] = (bf16)0.0f; }
  }
  __syncthreads();
  mlp_hidden(A0, 288, A1, S, W0, bi0, g0, be0, W1, bi1, g1, be1);
  v8f o[4] = {{}, {}, {}, {}};
  wmma_pass(A1, 64, 2, Wo, o);
  if (active) {
#pragma unroll
    for (int t = 0; t < 4; ++t) {
      const int nn = m + 16 * t; const float bb = bio[nn];
#pragma unroll
      for (int v = 0; v < 8; ++v)
        hbuf[(size_t)(n0 + v + 8 * hf) * 64 + nn] = o[t][v] + bb;
    }
  }
}

// ---------------------------------------------------------------------------
// final decoder: h [N,64] -> out [N,2]
// ---------------------------------------------------------------------------
__global__ __launch_bounds__(WPB * 32) void final_kernel(
    const float* __restrict__ hbuf, int N,
    const v16bf* W0, const float* bi0, const float* g0, const float* be0,
    const v16bf* W1, const float* bi1, const float* g1, const float* be1,
    const float* __restrict__ Wo, const float* __restrict__ bio,
    float* __restrict__ out) {
  __shared__ __align__(32) bf16 A0s[WPB][16 * 64];
  __shared__ __align__(32) bf16 A1s[WPB][16 * 64];
  __shared__ float Ss[WPB][16 * 64];
  const int wave = threadIdx.x >> 5, lane = threadIdx.x & 31;
  const int m = lane & 15, hf = lane >> 4;
  const int tile = blockIdx.x * WPB + wave;
  const bool active = tile * 16 < N;
  const int n0 = tile * 16;
  bf16* A0 = A0s[wave]; bf16* A1 = A1s[wave]; float* S = Ss[wave];
  if (active) {
    const float4* hp = (const float4*)(hbuf + (size_t)(n0 + m) * 64 + hf * 32);
    bf16* row = A0 + m * 64;
#pragma unroll
    for (int q = 0; q < 8; ++q) {
      const int j = hf * 32 + q * 4;
      float4 h4 = hp[q];
      row[j] = (bf16)h4.x; row[j + 1] = (bf16)h4.y;
      row[j + 2] = (bf16)h4.z; row[j + 3] = (bf16)h4.w;
    }
  }
  __syncthreads();
  mlp_hidden(A0, 64, A1, S, W0, bi0, g0, be0, W1, bi1, g1, be1);
  if (active && lane < 16) {
    float o0 = bio[0], o1 = bio[1];
    const bf16* r = A1 + m * 64;
    for (int j = 0; j < 64; ++j) {
      const float a = (float)r[j];
      o0 += a * Wo[2 * j];
      o1 += a * Wo[2 * j + 1];
    }
    out[(size_t)(n0 + m) * 2]     = o0;
    out[(size_t)(n0 + m) * 2 + 1] = o1;
  }
}

// ---------------------------------------------------------------------------
// host side
// ---------------------------------------------------------------------------
extern "C" void kernel_launch(void* const* d_in, const int* in_sizes, int n_in,
                              void* d_out, int out_size, void* d_ws, size_t ws_size,
                              hipStream_t stream) {
  (void)n_in; (void)out_size; (void)ws_size;
  const float* x   = (const float*)d_in[0];
  const float* pos = (const float*)d_in[1];
  const float* u   = (const float*)d_in[2];
  const int*   ei  = (const int*)d_in[3];
  // d_in[4] = batch (all zeros, single graph) -> ug broadcast, unused
  const int N = in_sizes[0] / 7;
  const int E = in_sizes[3] / 2;

  // workspace layout
  char* ws = (char*)d_ws;
  bf16* pack = (bf16*)ws;
  size_t off = 256 * 1024;
  float*    hbuf = (float*)(ws + off);    off += (size_t)N * 64 * 4;
  float*    sumb = (float*)(ws + off);    off += (size_t)N * 64 * 4;
  unsigned* mxb  = (unsigned*)(ws + off); off += (size_t)N * 64 * 4;
  float*    cntb = (float*)(ws + off);    off += (((size_t)N * 4) + 255) & ~(size_t)255;
  float*    ug   = (float*)(ws + off);    off += 256;
  float*    ea   = (float*)(ws + off);    // E*64*4 bytes

  // pack schedule: {d_in leaf index, fan_in, K padded to mult of 32}
  struct PM { int idx, fanin, kpad; };
  const PM pms[18] = {
    {5, 7, 32},     {9, 64, 64},                  // node_emb lin0, out
    {11, 3, 32},    {15, 64, 64},                 // edge_emb lin0, out
    {23, 64, 64},   {25, 64, 64},                 // final lin0, lin1
    {33, 208, 224}, {35, 64, 64}, {41, 64, 64},   // layer0 edge lin0, lin1, out
    {43, 272, 288}, {45, 64, 64}, {51, 64, 64},   // layer0 node
    {53, 208, 224}, {55, 64, 64}, {61, 64, 64},   // layer1 edge
    {63, 272, 288}, {65, 64, 64}, {71, 64, 64},   // layer1 node
  };
  size_t po[18];
  { size_t a = 0; for (int i = 0; i < 18; ++i) { po[i] = a; a += (size_t)pms[i].kpad * 64; } }
  for (int i = 0; i < 18; ++i)
    pack_kernel<<<(pms[i].kpad / 32) * 4, 32, 0, stream>>>(
        (const float*)d_in[pms[i].idx], pms[i].fanin, (v16bf*)(pack + po[i]));

#define PF(i) ((const float*)d_in[i])
#define PK(i) ((const v16bf*)(pack + po[i]))

  glob_kernel<<<1, 1, 0, stream>>>(u, PF(17), PF(18), PF(19), PF(20), PF(21), PF(22), ug);

  const int ntilesN = (N + 15) / 16, gridN = (ntilesN + WPB - 1) / WPB;
  const int ntilesE = (E + 15) / 16, gridE = (ntilesE + WPB - 1) / WPB;

  node_emb_kernel<<<gridN, WPB * 32, 0, stream>>>(
      x, N, hbuf, PK(0), PF(6), PF(7), PF(8), PK(1), PF(10));
  edge_emb_kernel<<<gridE, WPB * 32, 0, stream>>>(
      ei, E, pos, ea, PK(2), PF(12), PF(13), PF(14), PK(3), PF(16));

  for (int l = 0; l < 2; ++l) {
    const int eb = 33 + l * 20;   // edge-MLP leaf base
    const int nb = eb + 10;       // node-MLP leaf base
    const int pe = 6 + l * 6;     // pack base: edge
    const int pn = 9 + l * 6;     // pack base: node
    init_agg_kernel<<<(N * 64 + 255) / 256, 256, 0, stream>>>(sumb, mxb, cntb, N);
    edge_mlp_kernel<<<gridE, WPB * 32, 0, stream>>>(
        ei, E, hbuf, ea, ug,
        PK(pe),     PF(eb + 1), PF(eb + 4), PF(eb + 5),
        PK(pe + 1), PF(eb + 3), PF(eb + 6), PF(eb + 7),
        PK(pe + 2), PF(eb + 9),
        sumb, mxb, cntb);
    node_mlp_kernel<<<gridN, WPB * 32, 0, stream>>>(
        N, hbuf, sumb, mxb, cntb, ug,
        PK(pn),     PF(nb + 1), PF(nb + 4), PF(nb + 5),
        PK(pn + 1), PF(nb + 3), PF(nb + 6), PF(nb + 7),
        PK(pn + 2), PF(nb + 9));
  }

  final_kernel<<<gridN, WPB * 32, 0, stream>>>(
      hbuf, N,
      PK(4), PF(24), PF(27), PF(28),
      PK(5), PF(26), PF(29), PF(30),
      PF(31), PF(32), (float*)d_out);
#undef PF
#undef PK
}
